// FusedQKVAttention_Kernel_50508815401128
// MI455X (gfx1250) — matrix operand order repaired and checked by reading, not compiled, not run on hardware
//
#include <hip/hip_runtime.h>
#include <hip/hip_bf16.h>

// ---------------------------------------------------------------------------
// Fused QKV + multi-head attention for MI455X (gfx1250, wave32, WMMA).
//   B=2, N=2048, D=1024, H=16, Dh=64.  fp32 in/out, bf16 WMMA compute.
// Pipeline (workspace-staged, all on `stream`):
//   1) pack_x : x fp32 -> bf16                  [4096 x 1024]
//   2) pack_w : Wq/Wk/Wv fp32 -> bf16 TRANSPOSED (col-major) [3][1024][1024]
//   3) qkv_gemm: WMMA bf16 GEMM, 32x64 register tile per wave (8 WMMA/k-step)
//               Q scaled by 1/8; Q,K stored [B,H,N,64]; V stored transposed
//               [B,H,64,N] so the PV B-operand streams contiguously.
//   4) attn   : flash-attention, one wave = 32 queries (2 M-tiles) so K/V
//               fragments are reused; online softmax; P staged through LDS
//               to convert C-layout -> A-layout.
// ---------------------------------------------------------------------------

typedef __attribute__((ext_vector_type(16))) __bf16 v16bf;
typedef __attribute__((ext_vector_type(8)))  __bf16 v8bf;
typedef __attribute__((ext_vector_type(8)))  float  v8f;

#define EMBED   1024
#define NTOK    2048
#define BATCH   2
#define HEADS   16
#define DHEAD   64
#define MROWS   (BATCH * NTOK)        // 4096 tokens total

static __device__ inline v16bf cat8(v8bf a, v8bf b) {
  return __builtin_shufflevector(a, b, 0,1,2,3,4,5,6,7,8,9,10,11,12,13,14,15);
}

// 16-bit A operand 16x32 (MxK): lane holds row m = lane&15.
// element e in [0,8):  K = 8*half + e ; e in [8,16): K = 16 + 8*half + (e-8)
// -> two contiguous 16B loads per lane.
static __device__ inline v16bf load_a_frag(const __bf16* row, int hf) {
  v8bf c0 = *(const v8bf*)(row + hf * 8);
  v8bf c1 = *(const v8bf*)(row + 16 + hf * 8);
  return cat8(c0, c1);
}

// 16-bit B operand 32x16 (KxN): lane holds column n = lane&15.
// element e in [0,8): K = 8*half + e ; e in [8,16): K = 16 + 8*half + (e-8)
static __device__ inline v16bf load_b_frag(const __bf16* col, int hf) {
  v8bf c0 = *(const v8bf*)(col + hf * 8);
  v8bf c1 = *(const v8bf*)(col + 16 + hf * 8);
  return cat8(c0, c1);
}

static __device__ inline v8f wmma_bf16(v16bf a, v16bf b, v8f c) {
  // (neg_a, A, neg_b, B, c_mod, C, reuse_a, reuse_b)
  return __builtin_amdgcn_wmma_f32_16x16x32_bf16(false, a, false, b,
                                                 (short)0, c, false, false);
}

// ---------------------------------------------------------------------------
__global__ void pack_x_kernel(const float* __restrict__ x,
                              __bf16* __restrict__ xb, int n) {
  for (int i = blockIdx.x * blockDim.x + threadIdx.x; i < n;
       i += gridDim.x * blockDim.x)
    xb[i] = (__bf16)x[i];
}

// Wt[sel][c][k] = W_sel[k][c]  (bf16, column-major so B-fragments stream)
__global__ void pack_w_kernel(const float* __restrict__ Wq,
                              const float* __restrict__ Wk,
                              const float* __restrict__ Wv,
                              __bf16* __restrict__ Wt) {
  const int total = 3 * EMBED * EMBED;
  for (int i = blockIdx.x * blockDim.x + threadIdx.x; i < total;
       i += gridDim.x * blockDim.x) {
    int sel = i >> 20;
    int r   = i & ((1 << 20) - 1);      // r = c*1024 + k
    int c   = r >> 10;
    int k   = r & 1023;
    const float* W = (sel == 0) ? Wq : ((sel == 1) ? Wk : Wv);
    Wt[(size_t)sel * (EMBED * EMBED) + r] = (__bf16)W[k * EMBED + c];
  }
}

// ---------------------------------------------------------------------------
// One wave per 32x64 tile of [4096 x 1024] output: 2 A-frags x 4 B-frags
// -> 8 WMMAs per k-step with full fragment reuse.
// sel 0->Q (scaled 1/8), 1->K, 2->V(transposed store).
__global__ __launch_bounds__(256)
void qkv_gemm_kernel(const __bf16* __restrict__ xb,
                     const __bf16* __restrict__ Wt,
                     __bf16* __restrict__ Qb, __bf16* __restrict__ Kb,
                     __bf16* __restrict__ Vt) {
  const int wave = blockIdx.x * (blockDim.x >> 5) + (threadIdx.x >> 5);
  const int lane = threadIdx.x & 31;
  const int hf   = lane >> 4;
  const int l16  = lane & 15;

  // 3 mats x (4096/32) row-tiles x (1024/64) col-tiles = 3 * 128 * 16
  const int sel = wave / (128 * 16);
  const int rem = wave % (128 * 16);
  const int tm  = rem >> 4;          // 32-row tile (0..127)
  const int tc  = rem & 15;          // 64-col tile (0..15)

  const __bf16* arow0 = xb + (size_t)(tm * 32 + l16) * EMBED;
  const __bf16* arow1 = arow0 + (size_t)16 * EMBED;
  const __bf16* wbase = Wt + (size_t)sel * (EMBED * EMBED);
  const __bf16* bcol0 = wbase + (size_t)(tc * 64 +  0 + l16) * EMBED;
  const __bf16* bcol1 = wbase + (size_t)(tc * 64 + 16 + l16) * EMBED;
  const __bf16* bcol2 = wbase + (size_t)(tc * 64 + 32 + l16) * EMBED;
  const __bf16* bcol3 = wbase + (size_t)(tc * 64 + 48 + l16) * EMBED;

  v8f acc[2][4] = {};
  for (int k0 = 0; k0 < EMBED; k0 += 32) {
    const v16bf a0 = load_a_frag(arow0 + k0, hf);
    const v16bf a1 = load_a_frag(arow1 + k0, hf);
    const v16bf b0 = load_b_frag(bcol0 + k0, hf);
    const v16bf b1 = load_b_frag(bcol1 + k0, hf);
    const v16bf b2 = load_b_frag(bcol2 + k0, hf);
    const v16bf b3 = load_b_frag(bcol3 + k0, hf);
    acc[0][0] = wmma_bf16(a0, b0, acc[0][0]);
    acc[1][0] = wmma_bf16(a1, b0, acc[1][0]);
    acc[0][1] = wmma_bf16(a0, b1, acc[0][1]);
    acc[1][1] = wmma_bf16(a1, b1, acc[1][1]);
    acc[0][2] = wmma_bf16(a0, b2, acc[0][2]);
    acc[1][2] = wmma_bf16(a1, b2, acc[1][2]);
    acc[0][3] = wmma_bf16(a0, b3, acc[0][3]);
    acc[1][3] = wmma_bf16(a1, b3, acc[1][3]);
  }

  const float scale = (sel == 0) ? 0.125f : 1.0f;   // 1/sqrt(64) folded into Q
#pragma unroll
  for (int mt = 0; mt < 2; ++mt) {
#pragma unroll
    for (int d = 0; d < 4; ++d) {
#pragma unroll
      for (int r = 0; r < 8; ++r) {
        const int m  = tm * 32 + mt * 16 + r + 8 * hf;  // global token row
        const int c  = tc * 64 + d * 16 + l16;          // output channel
        const int bb = m >> 11, n = m & (NTOK - 1);
        const int hh = c >> 6,  dh = c & (DHEAD - 1);
        const size_t bhi = (size_t)(bb * HEADS + hh);
        const __bf16 v = (__bf16)(acc[mt][d][r] * scale);
        if (sel == 0)      Qb[(bhi * NTOK + n) * DHEAD + dh] = v;
        else if (sel == 1) Kb[(bhi * NTOK + n) * DHEAD + dh] = v;
        else               Vt[(bhi * DHEAD + dh) * NTOK + n] = v;  // transposed
      }
    }
  }
}

// ---------------------------------------------------------------------------
// Flash attention: one wave = 32 queries (two 16-row M tiles) of one (b,h).
// K and V fragments are loaded once per 32-key chunk and shared across both
// M tiles: 16 WMMAs per 16 global b128 loads.
__global__ __launch_bounds__(256)
void attn_kernel(const __bf16* __restrict__ Qb, const __bf16* __restrict__ Kb,
                 const __bf16* __restrict__ Vt, float* __restrict__ out) {
  __shared__ __align__(16) __bf16 lds[8][2][16 * 32];   // P staging per wave

  const int wv   = threadIdx.x >> 5;
  const int lane = threadIdx.x & 31;
  const int hf   = lane >> 4;
  const int l16  = lane & 15;
  const int wave = blockIdx.x * 8 + wv;
  const int bh   = wave >> 6;            // 0..31  (b*16+h)
  const int qt   = wave & 63;            // 32-query tile 0..63

  // Q fragments: 2 M tiles x 2 K-slices (Dh=64). Q pre-scaled by 1/8.
  v16bf aQ[2][2];
#pragma unroll
  for (int mt = 0; mt < 2; ++mt) {
    const __bf16* qrow =
        Qb + ((size_t)bh * NTOK + qt * 32 + mt * 16 + l16) * DHEAD;
    aQ[mt][0] = load_a_frag(qrow, hf);
    aQ[mt][1] = load_a_frag(qrow + 32, hf);
  }

  const v8f zero = {};
  v8f o[2][4] = {};
  float mi[2][8], li[2][8];
#pragma unroll
  for (int mt = 0; mt < 2; ++mt)
#pragma unroll
    for (int r = 0; r < 8; ++r) { mi[mt][r] = -__builtin_inff(); li[mt][r] = 0.0f; }

  const __bf16* Vbase = Vt + (size_t)bh * DHEAD * NTOK;

  for (int kb = 0; kb < NTOK; kb += 32) {
    // --- K fragments for this 32-key chunk (shared by both M tiles) ---
    const __bf16* k0row = Kb + ((size_t)bh * NTOK + kb + l16) * DHEAD;
    const __bf16* k1row = k0row + 16 * DHEAD;
    const v16bf bK0a = load_b_frag(k0row,      hf);
    const v16bf bK0b = load_b_frag(k0row + 32, hf);
    const v16bf bK1a = load_b_frag(k1row,      hf);
    const v16bf bK1b = load_b_frag(k1row + 32, hf);

#pragma unroll
    for (int mt = 0; mt < 2; ++mt) {
      // scores: two 16-key tiles, contraction over Dh=64
      v8f s0 = wmma_bf16(aQ[mt][0], bK0a, zero);
      s0     = wmma_bf16(aQ[mt][1], bK0b, s0);
      v8f s1 = wmma_bf16(aQ[mt][0], bK1a, zero);
      s1     = wmma_bf16(aQ[mt][1], bK1b, s1);

      // online softmax: row stats live across the 16 lanes of a half
      float alpha[8];
#pragma unroll
      for (int r = 0; r < 8; ++r) {
        float t = fmaxf(s0[r], s1[r]);
        t = fmaxf(t, __shfl_xor(t, 1, 32));
        t = fmaxf(t, __shfl_xor(t, 2, 32));
        t = fmaxf(t, __shfl_xor(t, 4, 32));
        t = fmaxf(t, __shfl_xor(t, 8, 32));
        const float mn = fmaxf(mi[mt][r], t);
        alpha[r] = __expf(mi[mt][r] - mn);
        const float p0 = __expf(s0[r] - mn);
        const float p1 = __expf(s1[r] - mn);
        float rs = p0 + p1;
        rs += __shfl_xor(rs, 1, 32);
        rs += __shfl_xor(rs, 2, 32);
        rs += __shfl_xor(rs, 4, 32);
        rs += __shfl_xor(rs, 8, 32);
        li[mt][r] = li[mt][r] * alpha[r] + rs;
        mi[mt][r] = mn;
        // stage P (C-layout -> LDS row-major 16x32)
        lds[wv][mt][(r + 8 * hf) * 32 + l16]      = (__bf16)p0;
        lds[wv][mt][(r + 8 * hf) * 32 + 16 + l16] = (__bf16)p1;
      }
#pragma unroll
      for (int r = 0; r < 8; ++r) {
        o[mt][0][r] *= alpha[r]; o[mt][1][r] *= alpha[r];
        o[mt][2][r] *= alpha[r]; o[mt][3][r] *= alpha[r];
      }
    }

    // within-wave DS ordering: make the P stores visible to the ds loads
    asm volatile("s_wait_dscnt 0" ::: "memory");

    // --- V fragments (shared by both M tiles), O += P x V ---
    const v16bf bV0 = load_b_frag(Vbase + (size_t)( 0 + l16) * NTOK + kb, hf);
    const v16bf bV1 = load_b_frag(Vbase + (size_t)(16 + l16) * NTOK + kb, hf);
    const v16bf bV2 = load_b_frag(Vbase + (size_t)(32 + l16) * NTOK + kb, hf);
    const v16bf bV3 = load_b_frag(Vbase + (size_t)(48 + l16) * NTOK + kb, hf);
#pragma unroll
    for (int mt = 0; mt < 2; ++mt) {
      const v16bf aP = load_a_frag(&lds[wv][mt][l16 * 32], hf);
      o[mt][0] = wmma_bf16(aP, bV0, o[mt][0]);
      o[mt][1] = wmma_bf16(aP, bV1, o[mt][1]);
      o[mt][2] = wmma_bf16(aP, bV2, o[mt][2]);
      o[mt][3] = wmma_bf16(aP, bV3, o[mt][3]);
    }
  }

  // --- normalize and store: out[b, tok, h*64 + dh] fp32 ---
  const int bb = bh >> 4, hh = bh & 15;
#pragma unroll
  for (int mt = 0; mt < 2; ++mt) {
#pragma unroll
    for (int r = 0; r < 8; ++r) {
      const float inv = 1.0f / li[mt][r];
      const int tok = qt * 32 + mt * 16 + r + 8 * hf;
      float* orow = out + ((size_t)bb * NTOK + tok) * EMBED + hh * DHEAD;
      orow[ 0 + l16] = o[mt][0][r] * inv;
      orow[16 + l16] = o[mt][1][r] * inv;
      orow[32 + l16] = o[mt][2][r] * inv;
      orow[48 + l16] = o[mt][3][r] * inv;
    }
  }
}

// ---------------------------------------------------------------------------
extern "C" void kernel_launch(void* const* d_in, const int* in_sizes, int n_in,
                              void* d_out, int out_size, void* d_ws,
                              size_t ws_size, hipStream_t stream) {
  const float* x  = (const float*)d_in[0];
  const float* Wq = (const float*)d_in[1];
  const float* Wk = (const float*)d_in[2];
  const float* Wv = (const float*)d_in[3];

  char* ws = (char*)d_ws;
  const size_t XB_BYTES = (size_t)MROWS * EMBED * 2;          // 8 MB
  const size_t WT_BYTES = (size_t)3 * EMBED * EMBED * 2;      // 6 MB
  const size_t QKV_BYTES = (size_t)BATCH * HEADS * NTOK * DHEAD * 2; // 8 MB

  __bf16* xb = (__bf16*)(ws);
  __bf16* wt = (__bf16*)(ws + XB_BYTES);
  __bf16* qb = (__bf16*)(ws + XB_BYTES + WT_BYTES);
  __bf16* kb = (__bf16*)(ws + XB_BYTES + WT_BYTES + QKV_BYTES);
  __bf16* vt = (__bf16*)(ws + XB_BYTES + WT_BYTES + 2 * QKV_BYTES);

  pack_x_kernel<<<1024, 256, 0, stream>>>(x, xb, MROWS * EMBED);
  pack_w_kernel<<<1024, 256, 0, stream>>>(Wq, Wk, Wv, wt);

  // 3 * 128 * 16 = 6144 tiles (32x64 each), 8 waves/block -> 768 blocks
  qkv_gemm_kernel<<<768, 256, 0, stream>>>(xb, wt, qb, kb, vt);

  // 2*16*64 = 2048 waves (32 queries each), 8 waves/block -> 256 blocks
  attn_kernel<<<256, 256, 0, stream>>>(qb, kb, vt, (float*)d_out);
}
